// GATNet_7043746365842
// MI455X (gfx1250) — compile-verified, matrix operand order checked by reading, not run on hardware
//
#include <hip/hip_runtime.h>
#include <hip/hip_bf16.h>

typedef _Float16 v16h __attribute__((ext_vector_type(16)));
typedef _Float16 h8   __attribute__((ext_vector_type(8)));
typedef float    v8f  __attribute__((ext_vector_type(8)));

#define LDA 136   // 128 + 8 halves padding; rows stay 16B-aligned (272B)

static __device__ __forceinline__ v16h frag16(const _Float16* p0, const _Float16* p1) {
    h8 a = *(const h8*)p0;
    h8 b = *(const h8*)p1;
    return __builtin_shufflevector(a, b, 0,1,2,3,4,5,6,7,8,9,10,11,12,13,14,15);
}

// monotonic float<->uint encoding so atomicMax(u32) == float max
static __device__ __forceinline__ unsigned fenc(float f) {
    unsigned u = __float_as_uint(f);
    return ((int)u < 0) ? ~u : (u | 0x80000000u);
}
static __device__ __forceinline__ float fdec(unsigned u) {
    return __uint_as_float(((int)u < 0) ? (u & 0x7fffffffu) : ~u);
}

// ---- BN scale/shift precompute: scale = gamma*rsqrt(var+eps); shift = beta - mean*scale
__global__ void bn_prep(const float* __restrict__ p, float* __restrict__ scale,
                        float* __restrict__ shift) {
    int f = threadIdx.x;              // 128 threads
    float g = p[f], b = p[128 + f], m = p[256 + f], v = p[384 + f];
    float s = g * rsqrtf(v + 1e-5f);
    scale[f] = s;
    shift[f] = b - m * s;
}

// ---- W[128x128] f32 row-major -> WT[c][k] f16 (transposed, so LDS B-frag loads are contiguous)
__global__ void conv_wt(const float* __restrict__ W, _Float16* __restrict__ WT) {
    int idx = blockIdx.x * 256 + threadIdx.x;   // 16384
    int c = idx >> 7, k = idx & 127;
    WT[idx] = (_Float16)W[k * 128 + c];
}

// ---- Y[N,128] = act( BN( pre(X) ) @ W + bias ), W given transposed f16.
//      pre(X) = PREB ? relu(X + preb[col]) : X      (fuses previous layer's bias+relu)
// One block = 128 rows x 128 cols, 8 waves; wave w owns column tile w.
template <int PREB, int BIAS, int RELU>
__global__ __launch_bounds__(256) void gemm_bn_wmma(
    const float* __restrict__ X, const float* __restrict__ scale,
    const float* __restrict__ shift, const _Float16* __restrict__ WT,
    const float* __restrict__ preb, const float* __restrict__ bias,
    float* __restrict__ Y, int n)
{
    __shared__ _Float16 sA[128 * LDA];
    __shared__ _Float16 sB[128 * LDA];
    const int t = threadIdx.x;
    const int blockRow = blockIdx.x * 128;

    // load B (already f16, transposed: WT[col][k])
    {
        int col = t >> 1, kb = (t & 1) * 64;
        const _Float16* src = WT + col * 128 + kb;
        _Float16* dst = sB + col * LDA + kb;
        #pragma unroll
        for (int j = 0; j < 64; j += 8) *(h8*)(dst + j) = *(const h8*)(src + j);
    }
    // load A rows with fused (bias+relu) + BatchNorm, f32 -> f16
    {
        int row = t >> 1, cb = (t & 1) * 64;
        int gr = blockRow + row;
        _Float16* dst = sA + row * LDA + cb;
        if (gr < n) {
            const float* src = X + (size_t)gr * 128 + cb;
            #pragma unroll
            for (int j = 0; j < 64; j += 4) {
                float4 v = *(const float4*)(src + j);
                int c = cb + j;
                if (PREB) {
                    v.x = fmaxf(v.x + preb[c + 0], 0.0f);
                    v.y = fmaxf(v.y + preb[c + 1], 0.0f);
                    v.z = fmaxf(v.z + preb[c + 2], 0.0f);
                    v.w = fmaxf(v.w + preb[c + 3], 0.0f);
                }
                dst[j + 0] = (_Float16)(v.x * scale[c + 0] + shift[c + 0]);
                dst[j + 1] = (_Float16)(v.y * scale[c + 1] + shift[c + 1]);
                dst[j + 2] = (_Float16)(v.z * scale[c + 2] + shift[c + 2]);
                dst[j + 3] = (_Float16)(v.w * scale[c + 3] + shift[c + 3]);
            }
        } else {
            #pragma unroll
            for (int j = 0; j < 64; ++j) dst[j] = (_Float16)0.0f;
        }
    }
    __syncthreads();

    const int wv = t >> 5, lane = t & 31;
    const int lrow = lane & 15;
    const int khalf = (lane >> 4) * 8;          // A/B fragment K split per ISA layout
    const int colL = wv * 16 + lrow;
    const bool fullBlock = (blockRow + 128 <= n);   // uniform per block

    v16h bf[4];
    #pragma unroll
    for (int kc = 0; kc < 4; ++kc) {
        const _Float16* bp = sB + colL * LDA + kc * 32 + khalf;
        bf[kc] = frag16(bp, bp + 16);
    }
    const float bv = BIAS ? bias[colL] : 0.0f;

    #pragma unroll
    for (int rt = 0; rt < 8; ++rt) {
        v8f acc = {};
        const _Float16* ap = sA + (rt * 16 + lrow) * LDA + khalf;
        #pragma unroll
        for (int kc = 0; kc < 4; ++kc) {
            v16h af = frag16(ap + kc * 32, ap + kc * 32 + 16);
            acc = __builtin_amdgcn_wmma_f32_16x16x32_f16(
                false, af, false, bf[kc], (short)0, acc, false, false);
        }
        const int rbase = blockRow + rt * 16 + 8 * (lane >> 4);
        float* yp = Y + (size_t)rbase * 128 + colL;
        if (fullBlock) {
            #pragma unroll
            for (int r = 0; r < 8; ++r) {
                float v = acc[r] + bv;
                if (RELU) v = fmaxf(v, 0.0f);
                yp[(size_t)r * 128] = v;
            }
        } else {
            #pragma unroll
            for (int r = 0; r < 8; ++r) {
                if (rbase + r < n) {
                    float v = acc[r] + bv;
                    if (RELU) v = fmaxf(v, 0.0f);
                    yp[(size_t)r * 128] = v;
                }
            }
        }
    }
}

// ---- per (node,head) attention logits: al = <hW[n,h,:], a[h,:]>
__global__ __launch_bounds__(256) void att_dots(
    const float* __restrict__ hW, const float* __restrict__ asrc,
    const float* __restrict__ adst, float* __restrict__ als,
    float* __restrict__ ald, int n)
{
    __shared__ float sa[128], sd[128];
    if (threadIdx.x < 128) { sa[threadIdx.x] = asrc[threadIdx.x]; sd[threadIdx.x] = adst[threadIdx.x]; }
    __syncthreads();
    int idx = blockIdx.x * 256 + threadIdx.x;
    if (idx >= n * 4) return;
    int node = idx >> 2, head = idx & 3;
    const float* hp = hW + (size_t)node * 128 + head * 32;
    float s = 0.f, d = 0.f;
    #pragma unroll 4
    for (int c = 0; c < 32; ++c) { float v = hp[c]; s += v * sa[head*32 + c]; d += v * sd[head*32 + c]; }
    als[idx] = s; ald[idx] = d;
}

// ---- pass 1: e = leaky_relu(al_s[src]+al_d[dst]); segment max via encoded atomicMax
__global__ __launch_bounds__(256) void edge_logits(
    const int* __restrict__ src, const int* __restrict__ dst,
    const float* __restrict__ als, const float* __restrict__ ald,
    float* __restrict__ ebuf, unsigned* __restrict__ menc, int E, int ET)
{
    int idx = blockIdx.x * 256 + threadIdx.x;
    if (idx >= ET * 4) return;
    int e = idx >> 2, h = idx & 3;
    int s, d;
    if (e < E) { s = src[e]; d = dst[e]; } else { s = d = e - E; }
    float v = als[s * 4 + h] + ald[d * 4 + h];
    v = v > 0.f ? v : 0.2f * v;
    ebuf[idx] = v;
    atomicMax(menc + d * 4 + h, fenc(v));
}

// ---- pass 2: w = exp(e - max[dst]); segment sum of w
__global__ __launch_bounds__(256) void edge_expsum(
    const int* __restrict__ dst, float* __restrict__ ebuf,
    const unsigned* __restrict__ menc, float* __restrict__ denom, int E, int ET)
{
    int idx = blockIdx.x * 256 + threadIdx.x;
    if (idx >= ET * 4) return;
    int e = idx >> 2, h = idx & 3;
    int d = (e < E) ? dst[e] : e - E;
    float w = __expf(ebuf[idx] - fdec(menc[d * 4 + h]));
    ebuf[idx] = w;
    atomicAdd(denom + d * 4 + h, w);
}

// ---- pass 3: wave per edge; lane l handles feats [4l,4l+4); scatter-add alpha*h[src]
__global__ __launch_bounds__(256) void edge_scatter(
    const int* __restrict__ src, const int* __restrict__ dst,
    const float* __restrict__ hW, const float* __restrict__ ebuf,
    const float* __restrict__ denom, float* __restrict__ accum, int E, int ET)
{
    int lane = threadIdx.x & 31;
    int e = blockIdx.x * 8 + (threadIdx.x >> 5);
    if (e >= ET) return;
    int s, d;
    if (e < E) { s = src[e]; d = dst[e]; } else { s = d = e - E; }
    int f = lane * 4, h = f >> 5;
    float alpha = ebuf[e * 4 + h] / (denom[d * 4 + h] + 1e-16f);
    float4 hv = *(const float4*)(hW + (size_t)s * 128 + f);
    float* ap = accum + (size_t)d * 128 + f;
    atomicAdd(ap + 0, hv.x * alpha);
    atomicAdd(ap + 1, hv.y * alpha);
    atomicAdd(ap + 2, hv.z * alpha);
    atomicAdd(ap + 3, hv.w * alpha);
}

// ---- global_add_pool with fused final-layer bias+relu
__global__ __launch_bounds__(256) void pool_sum(
    const float* __restrict__ h, const float* __restrict__ bias,
    const int* __restrict__ batch, float* __restrict__ g, int total)
{
    int idx = blockIdx.x * 256 + threadIdx.x;
    if (idx >= total) return;
    int node = idx >> 7, f = idx & 127;
    float v = fmaxf(h[idx] + bias[f], 0.0f);
    atomicAdd(g + (size_t)batch[node] * 128 + f, v);
}

// ---- logits[gi,c] = BN(g[gi]) . cls_W[:,c] + cls_b[c]
__global__ __launch_bounds__(256) void classify(
    const float* __restrict__ g, const float* __restrict__ scale,
    const float* __restrict__ shift, const float* __restrict__ clsW,
    const float* __restrict__ clsb, float* __restrict__ logits, int G)
{
    int idx = blockIdx.x * 256 + threadIdx.x;
    if (idx >= G * 10) return;
    int gi = idx / 10, c = idx % 10;
    float acc = clsb[c];
    for (int f = 0; f < 128; ++f)
        acc += (g[gi * 128 + f] * scale[f] + shift[f]) * clsW[f * 10 + c];
    logits[idx] = acc;
}

__global__ __launch_bounds__(256) void log_softmax_k(
    const float* __restrict__ logits, float* __restrict__ out, int G)
{
    int gi = blockIdx.x * 256 + threadIdx.x;
    if (gi >= G) return;
    const float* l = logits + gi * 10;
    float m = l[0];
    for (int c = 1; c < 10; ++c) m = fmaxf(m, l[c]);
    float s = 0.f;
    for (int c = 0; c < 10; ++c) s += __expf(l[c] - m);
    float ls = logf(s);
    for (int c = 0; c < 10; ++c) out[gi * 10 + c] = l[c] - m - ls;
}

extern "C" void kernel_launch(void* const* d_in, const int* in_sizes, int n_in,
                              void* d_out, int out_size, void* d_ws, size_t ws_size,
                              hipStream_t stream) {
    const float* x        = (const float*)d_in[0];
    const float* bn_feat  = (const float*)d_in[1];
    const float* feat_W   = (const float*)d_in[2];
    const float* feat_b   = (const float*)d_in[3];
    const float* bns_conv = (const float*)d_in[4];
    const float* gat_W    = (const float*)d_in[5];
    const float* att_src  = (const float*)d_in[6];
    const float* att_dst  = (const float*)d_in[7];
    const float* gat_bias = (const float*)d_in[8];
    const float* bn_hid   = (const float*)d_in[9];
    const float* cls_W    = (const float*)d_in[10];
    const float* cls_b    = (const float*)d_in[11];
    const int*   eidx     = (const int*)d_in[12];
    const int*   batch    = (const int*)d_in[13];
    float* out = (float*)d_out;

    const int Nn = in_sizes[13];
    const int Ee = in_sizes[12] / 2;
    const int ET = Ee + Nn;
    const int Gg = out_size / 10;
    const int Lc = 3;
    const int* srcA = eidx;
    const int* dstA = eidx + Ee;

    // carve workspace
    char* p = (char*)d_ws;
    auto carve = [&](size_t bytes) { char* r = p; p += (bytes + 255) & ~(size_t)255; return (void*)r; };
    float*     scaleB = (float*)carve(5 * 128 * sizeof(float));
    float*     shiftB = (float*)carve(5 * 128 * sizeof(float));
    _Float16*  WT     = (_Float16*)carve(4 * 128 * 128 * sizeof(_Float16));
    float*     hA     = (float*)carve((size_t)Nn * 128 * sizeof(float));
    float*     hB     = (float*)carve((size_t)Nn * 128 * sizeof(float));
    float*     hWb    = (float*)carve((size_t)Nn * 128 * sizeof(float));
    float*     als    = (float*)carve((size_t)Nn * 4 * sizeof(float));
    float*     ald    = (float*)carve((size_t)Nn * 4 * sizeof(float));
    unsigned*  menc   = (unsigned*)carve((size_t)Nn * 4 * sizeof(unsigned));
    float*     denom  = (float*)carve((size_t)Nn * 4 * sizeof(float));
    float*     ebuf   = (float*)carve((size_t)ET * 4 * sizeof(float));
    float*     gpool  = (float*)carve((size_t)Gg * 128 * sizeof(float));
    float*     logits = (float*)carve((size_t)Gg * 10 * sizeof(float));

    // BN constants: slot0 = feat, 1..3 = conv layers, 4 = hidden
    bn_prep<<<1, 128, 0, stream>>>(bn_feat, scaleB, shiftB);
    for (int i = 0; i < Lc; ++i)
        bn_prep<<<1, 128, 0, stream>>>(bns_conv + (size_t)i * 512,
                                       scaleB + (1 + i) * 128, shiftB + (1 + i) * 128);
    bn_prep<<<1, 128, 0, stream>>>(bn_hid, scaleB + 4 * 128, shiftB + 4 * 128);

    // weights -> transposed f16
    conv_wt<<<64, 256, 0, stream>>>(feat_W, WT);
    for (int i = 0; i < Lc; ++i)
        conv_wt<<<64, 256, 0, stream>>>(gat_W + (size_t)i * 16384, WT + (size_t)(1 + i) * 16384);

    const int gemmGrid = (Nn + 127) / 128;
    const int nhGrid   = (Nn * 4 + 255) / 256;
    const int ehGrid   = (ET * 4 + 255) / 256;
    const int elGrid   = (ET + 7) / 8;
    const int nfGrid   = (Nn * 128 + 255) / 256;

    // h = relu(BN(x) @ feat_W + feat_b)
    gemm_bn_wmma<0, 1, 1><<<gemmGrid, 256, 0, stream>>>(
        x, scaleB, shiftB, WT, nullptr, feat_b, hA, Nn);

    const float* curIn   = hA;       // current node features (raw; may need preb+relu)
    const float* curPreb = nullptr;  // pending bias+relu to apply before BN
    float*       accum   = hB;       // aggregation target this layer

    for (int i = 0; i < Lc; ++i) {
        const float* sc = scaleB + (1 + i) * 128;
        const float* sh = shiftB + (1 + i) * 128;
        const _Float16* wt = WT + (size_t)(1 + i) * 16384;

        // hW = BN( pre(h) ) @ gat_W[i]
        if (curPreb)
            gemm_bn_wmma<1, 0, 0><<<gemmGrid, 256, 0, stream>>>(
                curIn, sc, sh, wt, curPreb, nullptr, hWb, Nn);
        else
            gemm_bn_wmma<0, 0, 0><<<gemmGrid, 256, 0, stream>>>(
                curIn, sc, sh, wt, nullptr, nullptr, hWb, Nn);

        att_dots<<<nhGrid, 256, 0, stream>>>(
            hWb, att_src + (size_t)i * 128, att_dst + (size_t)i * 128, als, ald, Nn);

        hipMemsetAsync(menc,  0, (size_t)Nn * 4 * sizeof(unsigned), stream);
        hipMemsetAsync(denom, 0, (size_t)Nn * 4 * sizeof(float), stream);
        hipMemsetAsync(accum, 0, (size_t)Nn * 128 * sizeof(float), stream);

        edge_logits<<<ehGrid, 256, 0, stream>>>(srcA, dstA, als, ald, ebuf, menc, Ee, ET);
        edge_expsum<<<ehGrid, 256, 0, stream>>>(dstA, ebuf, menc, denom, Ee, ET);
        edge_scatter<<<elGrid, 256, 0, stream>>>(srcA, dstA, hWb, ebuf, denom, accum, Ee, ET);

        // bias+relu deferred: fused into next consumer (GEMM A-loader or pool)
        curPreb = gat_bias + (size_t)i * 128;
        float* next = (accum == hB) ? hA : hB;
        curIn = accum;
        accum = next;
    }

    hipMemsetAsync(gpool, 0, (size_t)Gg * 128 * sizeof(float), stream);
    pool_sum<<<nfGrid, 256, 0, stream>>>(curIn, curPreb, batch, gpool, Nn * 128);

    classify<<<(Gg * 10 + 255) / 256, 256, 0, stream>>>(
        gpool, scaleB + 4 * 128, shiftB + 4 * 128, cls_W, cls_b, logits, Gg);
    log_softmax_k<<<(Gg + 255) / 256, 256, 0, stream>>>(logits, out, Gg);
}
